// PatchMTSC_32736240730551
// MI455X (gfx1250) — matrix operand order repaired and checked by reading, not compiled, not run on hardware
//
#include <hip/hip_runtime.h>

#define BN_EPS 1e-5f
#define SLOPE 0.01f

typedef __bf16 bf16;
typedef __attribute__((ext_vector_type(16))) __bf16 v16bf;
typedef __attribute__((ext_vector_type(8)))  __bf16 v8bf;
typedef __attribute__((ext_vector_type(8)))  float  v8f;
typedef unsigned int v4u __attribute__((ext_vector_type(4)));
typedef unsigned int v8u __attribute__((ext_vector_type(8)));

__device__ __forceinline__ float lrelu(float x) { return x >= 0.f ? x : SLOPE * x; }

__device__ __forceinline__ v8f vzero8() {
  v8f r;
#pragma unroll
  for (int i = 0; i < 8; ++i) r[i] = 0.f;
  return r;
}

__device__ __forceinline__ v8f wmma_bf16(v16bf a, v16bf b, v8f c) {
  return __builtin_amdgcn_wmma_f32_16x16x32_bf16(false, a, false, b, (short)0, c,
                                                 false, false);
}

// A-matrix fragment (16x32 bf16): lane holds row M=l15, K chunks at
// kbase + halfA + {0..7} and kbase + 16 + halfA + {0..7}   (halfA = (lane>>4)*8)
__device__ __forceinline__ v16bf ldfragA(const bf16* row, int kbase, int halfA) {
  v8bf lo = *(const v8bf*)(row + kbase + halfA);
  v8bf hi = *(const v8bf*)(row + kbase + 16 + halfA);
  v16bf r;
#pragma unroll
  for (int i = 0; i < 8; ++i) { r[i] = lo[i]; r[i + 8] = hi[i]; }
  return r;
}

// B-matrix fragment (32x16 bf16): lane holds col N=l15, K chunks at
// kbase + halfB/2 + {0..7} and kbase + 16 + halfB/2 + {0..7}   (halfB = (lane>>4)*16)
__device__ __forceinline__ v16bf ldfragB(const bf16* row, int kbase, int halfB) {
  v8bf lo = *(const v8bf*)(row + kbase + (halfB >> 1));
  v8bf hi = *(const v8bf*)(row + kbase + 16 + (halfB >> 1));
  v16bf r;
#pragma unroll
  for (int i = 0; i < 8; ++i) { r[i] = lo[i]; r[i + 8] = hi[i]; }
  return r;
}

// ---------------- stats / BN helpers ----------------

__global__ void k_zero(float* p) {
  int i = threadIdx.x;
#pragma unroll
  for (int k = 0; k < 4; ++k) p[i + 256 * k] = 0.f;
}

__global__ __launch_bounds__(256) void k_xstats(const float* __restrict__ x,
                                                float* __restrict__ gsum,
                                                float* __restrict__ gsq) {
  __shared__ float ssum[128], ssq[128];
  int tid = threadIdx.x;
  if (tid < 128) { ssum[tid] = 0.f; ssq[tid] = 0.f; }
  __syncthreads();
  size_t base = (size_t)blockIdx.x * 16384;
  float s1 = 0.f, s2 = 0.f;
#pragma unroll 4
  for (int k = 0; k < 64; ++k) {
    float v = x[base + tid + 256 * k];  // channel = (tid + 256k) & 127 = tid & 127
    s1 += v; s2 += v * v;
  }
  atomicAdd(&ssum[tid & 127], s1);
  atomicAdd(&ssq[tid & 127], s2);
  __syncthreads();
  if (tid < 128) { atomicAdd(&gsum[tid], ssum[tid]); atomicAdd(&gsq[tid], ssq[tid]); }
}

__global__ void k_finalize(const float* __restrict__ gsum, const float* __restrict__ gsq,
                           const float* __restrict__ gamma, const float* __restrict__ beta,
                           float* __restrict__ a, float* __restrict__ c, float inv_n) {
  int i = threadIdx.x;
  float mean = gsum[i] * inv_n;
  float var  = gsq[i] * inv_n - mean * mean;
  float s = gamma[i] * rsqrtf(var + BN_EPS);
  a[i] = s;
  c[i] = beta[i] - mean * s;
}

// ---------------- nf = bf16(x @ W_map + b_map), xb = bf16(BN(x)) ----------------

__global__ __launch_bounds__(256) void k_map(const float* __restrict__ x,
                                             const float* __restrict__ Wmap,
                                             const float* __restrict__ bmap,
                                             const float* __restrict__ a_in,
                                             const float* __restrict__ c_in,
                                             bf16* __restrict__ nf,
                                             bf16* __restrict__ xb) {
  __shared__ __align__(16) bf16 sX[128 * 128];    // x tile, row-major [row][k]
  __shared__ __align__(16) bf16 sWt[128 * 128];   // W_map transposed [n][k]
  int tid = threadIdx.x;
  size_t base = (size_t)blockIdx.x * 16384;
  for (int i = tid; i < 16384; i += 256) {
    sWt[(i & 127) * 128 + (i >> 7)] = (bf16)Wmap[i];
    float xv = x[base + i];
    sX[i] = (bf16)xv;
    int ch = i & 127;
    xb[base + i] = (bf16)(a_in[ch] * xv + c_in[ch]);
  }
  __syncthreads();
  int w = tid >> 5, lane = tid & 31, l15 = lane & 15;
  int halfA = (lane >> 4) * 8, halfB = (lane >> 4) * 16;
  int tr = w * 16;
  v16bf afr[4];
#pragma unroll
  for (int kk = 0; kk < 4; ++kk) afr[kk] = ldfragA(sX + (tr + l15) * 128, kk * 32, halfA);
  size_t row0 = (size_t)blockIdx.x * 128 + tr;
#pragma unroll
  for (int nt = 0; nt < 8; ++nt) {
    v8f acc = vzero8();
#pragma unroll
    for (int kk = 0; kk < 4; ++kk) {
      v16bf bfr = ldfragB(sWt + (nt * 16 + l15) * 128, kk * 32, halfB);
      acc = wmma_bf16(afr[kk], bfr, acc);
    }
    int col = nt * 16 + l15;
    float bias = bmap[col];
#pragma unroll
    for (int v = 0; v < 8; ++v)
      nf[(row0 + v + halfA) * 128 + col] = (bf16)(acc[v] + bias);
  }
}

// ---------------- fused: scores -> online softmax -> (softmax*pre + eye*pre) @ xb ----------------

__global__ __launch_bounds__(512) void k_fused(const bf16* __restrict__ nf,
                                               const bf16* __restrict__ xb,
                                               const float* __restrict__ pre,
                                               float* __restrict__ hmid) {
  __shared__ __align__(16) bf16 sNF[512 * 128];   // nf_b row-major         128 KB
  __shared__ __align__(16) bf16 sXT[128 * 512];   // xb_b transposed [d][n] 128 KB
  __shared__ __align__(16) bf16 sP[16 * 512];     // per-wave P staging      16 KB
  int b = blockIdx.x >> 1;
  int split = blockIdx.x & 1;
  int tid = threadIdx.x;

  // --- TDM bulk copy: nf_b (131072 B) -> sNF as one 1-D tensor tile ---
  // D# group0: count=1, lds_addr, 57-bit global_addr, type=2 ("image")
  // D# group1: data_size=3 (8B), tensor_dim0 = tile_dim0 = 16384, dim0_stride = 16384
  if (tid < 32) {
    unsigned long long ga = (unsigned long long)(nf + (size_t)b * 65536);
    unsigned int lds_off = (unsigned int)(unsigned long long)(void*)sNF;
    v4u g0;
    g0[0] = 1u;                                      // count=1, user mode
    g0[1] = lds_off;                                 // lds_addr
    g0[2] = (unsigned int)ga;                        // global_addr[31:0]
    g0[3] = 0x80000000u |                            // type=2 in bits[127:126]
            ((unsigned int)(ga >> 32) & 0x01FFFFFFu);// global_addr[56:32]
    v8u g1;
    g1[0] = 0x00030000u;   // workgroup_mask=0, data_size=3 (8 bytes)
    g1[1] = 0x40000000u;   // tensor_dim0[15:0]=16384 in bits[63:48]
    g1[2] = 0u;            // tensor_dim0[31:16]=0, tensor_dim1=0 (unused)
    g1[3] = 0x40000000u;   // tile_dim0=16384 in bits[127:112]
    g1[4] = 0u;            // tile_dim1=0, tile_dim2=0 (1-D)
    g1[5] = 16384u;        // tensor_dim0_stride[31:0]
    g1[6] = 0u;
    g1[7] = 0u;
    asm volatile("tensor_load_to_lds %0, %1" :: "s"(g0), "s"(g1) : "memory");
    __builtin_amdgcn_s_wait_tensorcnt(0);
  }
  {  // load xb_b transposed into LDS (TDM cannot transpose 16-bit data)
    const bf16* g = xb + (size_t)b * 65536;
#pragma unroll
    for (int ck = 0; ck < 16; ++ck) {
      int e = (tid + ck * 512) * 8;
      v8bf v = *(const v8bf*)(g + e);
      int n = e >> 7, d0 = e & 127;
#pragma unroll
      for (int j = 0; j < 8; ++j) sXT[(d0 + j) * 512 + n] = v[j];
    }
  }
  __syncthreads();
  int w = tid >> 5, lane = tid & 31, l15 = lane & 15;
  int halfA = (lane >> 4) * 8, halfB = (lane >> 4) * 16;
  int r0 = split * 256 + w * 16;
  v16bf afr[4];
#pragma unroll
  for (int kk = 0; kk < 4; ++kk) afr[kk] = ldfragA(sNF + (r0 + l15) * 128, kk * 32, halfA);
  float mrow[8], lrow[8];
  v8f O[8];
#pragma unroll
  for (int v = 0; v < 8; ++v) { mrow[v] = -3.0e38f; lrow[v] = 0.f; }
#pragma unroll
  for (int nt = 0; nt < 8; ++nt) O[nt] = vzero8();
  bf16* myP = sP + w * 512;  // 16 x 32 bf16 per wave

  for (int jj = 0; jj < 16; ++jj) {  // 32 adj-columns per step
    v8f s0 = vzero8(), s1 = vzero8();
#pragma unroll
    for (int kk = 0; kk < 4; ++kk) {
      v16bf b0 = ldfragB(sNF + (jj * 32 + l15) * 128, kk * 32, halfB);
      v16bf b1 = ldfragB(sNF + (jj * 32 + 16 + l15) * 128, kk * 32, halfB);
      s0 = wmma_bf16(afr[kk], b0, s0);
      s1 = wmma_bf16(afr[kk], b1, s1);
    }
    float scl[8];
#pragma unroll
    for (int v = 0; v < 8; ++v) {
      int grow = r0 + v + halfA;
      int gc0 = jj * 32 + l15, gc1 = gc0 + 16;
      float v0 = s0[v]; if (grow == gc0) v0 -= 1.0e8f; v0 = lrelu(v0);
      float v1 = s1[v]; if (grow == gc1) v1 -= 1.0e8f; v1 = lrelu(v1);
      float rm = fmaxf(v0, v1);
      rm = fmaxf(rm, __shfl_xor(rm, 1, 16));
      rm = fmaxf(rm, __shfl_xor(rm, 2, 16));
      rm = fmaxf(rm, __shfl_xor(rm, 4, 16));
      rm = fmaxf(rm, __shfl_xor(rm, 8, 16));
      float mn = fmaxf(mrow[v], rm);
      float sc = __expf(mrow[v] - mn);
      float e0 = __expf(v0 - mn), e1 = __expf(v1 - mn);
      float rs = e0 + e1;
      rs += __shfl_xor(rs, 1, 16);
      rs += __shfl_xor(rs, 2, 16);
      rs += __shfl_xor(rs, 4, 16);
      rs += __shfl_xor(rs, 8, 16);
      lrow[v] = lrow[v] * sc + rs;
      mrow[v] = mn;
      scl[v] = sc;
      // pre_relation multiplies post-softmax probabilities (all-ones, applied faithfully)
      float p0 = e0 * pre[(size_t)grow * 512 + gc0];
      float p1 = e1 * pre[(size_t)grow * 512 + gc1];
      int prow = v + halfA;
      myP[prow * 32 + l15]      = (bf16)p0;
      myP[prow * 32 + 16 + l15] = (bf16)p1;
    }
#pragma unroll
    for (int nt = 0; nt < 8; ++nt) {
      v8f o = O[nt];
#pragma unroll
      for (int v = 0; v < 8; ++v) o[v] *= scl[v];
      O[nt] = o;
    }
    // per-wave LDS RAW: DS ops are in-order within a wave, no barrier needed
    v16bf pA = ldfragA(myP + l15 * 32, 0, halfA);
#pragma unroll
    for (int nt = 0; nt < 8; ++nt) {
      v16bf xf = ldfragB(sXT + (nt * 16 + l15) * 512, jj * 32, halfB);
      O[nt] = wmma_bf16(pA, xf, O[nt]);
    }
  }
  // normalize + eye*pre diagonal term, write f32 hmid
  float* hout = hmid + (size_t)b * 65536;
#pragma unroll
  for (int v = 0; v < 8; ++v) {
    int grow = r0 + v + halfA;
    float inv = 1.0f / lrow[v];
    float pd = pre[(size_t)grow * 512 + grow];
#pragma unroll
    for (int nt = 0; nt < 8; ++nt) {
      int d = nt * 16 + l15;
      float xv = (float)sXT[d * 512 + grow];
      hout[(size_t)grow * 128 + d] = O[nt][v] * inv + pd * xv;
    }
  }
}

// ---------------- h2 = hmid @ W_theta + b_theta (pre-BN, into d_out) + channel stats ----------------

__global__ __launch_bounds__(256) void k_gemm2(const float* __restrict__ hmid,
                                               const float* __restrict__ Wt,
                                               const float* __restrict__ bt,
                                               float* __restrict__ out,
                                               float* __restrict__ gsum,
                                               float* __restrict__ gsq) {
  __shared__ __align__(16) bf16 sH[128 * 128];
  __shared__ __align__(16) bf16 sWt[128 * 128];  // W_theta transposed [n][k]
  __shared__ float ssum[128], ssq[128];
  int tid = threadIdx.x;
  if (tid < 128) { ssum[tid] = 0.f; ssq[tid] = 0.f; }
  size_t base = (size_t)blockIdx.x * 16384;
  for (int i = tid; i < 16384; i += 256) {
    sWt[(i & 127) * 128 + (i >> 7)] = (bf16)Wt[i];
    sH[i] = (bf16)hmid[base + i];
  }
  __syncthreads();
  int w = tid >> 5, lane = tid & 31, l15 = lane & 15;
  int halfA = (lane >> 4) * 8, halfB = (lane >> 4) * 16;
  int tr = w * 16;
  v16bf afr[4];
#pragma unroll
  for (int kk = 0; kk < 4; ++kk) afr[kk] = ldfragA(sH + (tr + l15) * 128, kk * 32, halfA);
  size_t row0 = (size_t)blockIdx.x * 128 + tr;
#pragma unroll
  for (int nt = 0; nt < 8; ++nt) {
    v8f acc = vzero8();
#pragma unroll
    for (int kk = 0; kk < 4; ++kk) {
      v16bf bfr = ldfragB(sWt + (nt * 16 + l15) * 128, kk * 32, halfB);
      acc = wmma_bf16(afr[kk], bfr, acc);
    }
    int col = nt * 16 + l15;
    float bias = bt[col];
    float s1 = 0.f, s2 = 0.f;
#pragma unroll
    for (int v = 0; v < 8; ++v) {
      float val = acc[v] + bias;
      out[(row0 + v + halfA) * 128 + col] = val;
      s1 += val; s2 += val * val;
    }
    atomicAdd(&ssum[col], s1);
    atomicAdd(&ssq[col], s2);
  }
  __syncthreads();
  if (tid < 128) { atomicAdd(&gsum[tid], ssum[tid]); atomicAdd(&gsq[tid], ssq[tid]); }
}

__global__ void k_apply(float* __restrict__ out, const float* __restrict__ a,
                        const float* __restrict__ c) {
  size_t i = (size_t)blockIdx.x * 256 + threadIdx.x;
  int ch = (int)(i & 127);
  out[i] = lrelu(a[ch] * out[i] + c[ch]);
}

// ---------------- launch ----------------

extern "C" void kernel_launch(void* const* d_in, const int* in_sizes, int n_in,
                              void* d_out, int out_size, void* d_ws, size_t ws_size,
                              hipStream_t stream) {
  const float* x      = (const float*)d_in[0];
  const float* Wmap   = (const float*)d_in[1];
  const float* bmap   = (const float*)d_in[2];
  const float* Wtheta = (const float*)d_in[3];
  const float* btheta = (const float*)d_in[4];
  const float* gin    = (const float*)d_in[5];
  const float* bin    = (const float*)d_in[6];
  const float* gout   = (const float*)d_in[7];
  const float* bout   = (const float*)d_in[8];
  const float* pre    = (const float*)d_in[9];
  float* out = (float*)d_out;

  float* wsf    = (float*)d_ws;
  float* gsum_x = wsf +   0; float* gsq_x = wsf + 128;
  float* a_in   = wsf + 256; float* c_in  = wsf + 384;
  float* gsum_h = wsf + 512; float* gsq_h = wsf + 640;
  float* a_out  = wsf + 768; float* c_out = wsf + 896;
  bf16*  nf   = (bf16*)(wsf + 1024);          // 8,388,608 bf16 (16 MB)
  bf16*  xbuf = nf + (size_t)8388608;         // 8,388,608 bf16 (16 MB)
  float* hmid = (float*)(xbuf + (size_t)8388608);  // 8,388,608 f32 (32 MB)

  k_zero<<<1, 256, 0, stream>>>(wsf);
  k_xstats<<<512, 256, 0, stream>>>(x, gsum_x, gsq_x);
  k_finalize<<<1, 128, 0, stream>>>(gsum_x, gsq_x, gin, bin, a_in, c_in, 1.0f / 65536.0f);
  k_map<<<512, 256, 0, stream>>>(x, Wmap, bmap, a_in, c_in, nf, xbuf);
  k_fused<<<256, 512, 0, stream>>>(nf, xbuf, pre, hmid);
  k_gemm2<<<512, 256, 0, stream>>>(hmid, Wtheta, btheta, out, gsum_h, gsq_h);
  k_finalize<<<1, 128, 0, stream>>>(gsum_h, gsq_h, gout, bout, a_out, c_out, 1.0f / 65536.0f);
  k_apply<<<32768, 256, 0, stream>>>(out, a_out, c_out);
  (void)in_sizes; (void)n_in; (void)out_size; (void)ws_size;
}
